// GNNEncoder_57964878627397
// MI455X (gfx1250) — matrix operand order repaired and checked by reading, not compiled, not run on hardware
//
#include <hip/hip_runtime.h>
#include <hip/hip_bf16.h>

#define NNODES 10000
#define NEDGES 160000
#define HD 128

typedef __attribute__((ext_vector_type(16))) _Float16 v16h;
typedef __attribute__((ext_vector_type(8)))  float    v8f;

__device__ __forceinline__ float silu_f(float x){ return x / (1.0f + __expf(-x)); }
__device__ __forceinline__ float elu_f(float x){ return x > 0.0f ? x : (__expf(x) - 1.0f); }

// ---------------- WMMA fragment helpers (V_WMMA_F32_16X16X32_F16) ----------------
// A (16x32 f16): lane l: M = l%16; VGPR v: K = (v>=4?16:0) + (l>=16?8:0) + 2*(v%4) + h
__device__ __forceinline__ v16h a_frag_h(const _Float16* base, int stride, int k0, int lane){
  const _Float16* row = base + (lane & 15) * stride;
  const int kb = k0 + ((lane & 16) ? 8 : 0);
  union { v16h v; unsigned u[8]; } r;
#pragma unroll
  for (int vv = 0; vv < 8; ++vv){
    const int k = kb + ((vv & 4) << 2) + ((vv & 3) << 1);   // always even
    __builtin_memcpy(&r.u[vv], row + k, 4);
  }
  return r.v;
}

// A fragment built from an f32 LDS tile (convert on load)
__device__ __forceinline__ v16h a_frag_f(const float* base, int stride, int k0, int lane){
  const float* row = base + (lane & 15) * stride;
  const int kb = k0 + ((lane & 16) ? 8 : 0);
  v16h a;
#pragma unroll
  for (int vv = 0; vv < 8; ++vv){
    const int k = kb + ((vv & 4) << 2) + ((vv & 3) << 1);
    a[2*vv]   = (_Float16)row[k];
    a[2*vv+1] = (_Float16)row[k+1];
  }
  return a;
}

// B (32x16 f16): lane l: N = l%16; lanes 0-15 hold K=0..7,16..23, lanes 16-31 hold K=8..15,24..31
// stored pre-swizzled: [((kstep*8 + nt)*32 + lane)*16 + i]
__device__ __forceinline__ v16h b_frag(const _Float16* wswz, int kstep, int nt, int lane){
  v16h b;
  __builtin_memcpy(&b, wswz + (((size_t)(kstep*8 + nt)*32 + lane)*16), 32);
  return b;
}

// Fill pre-swizzled B storage from global f32 weights W[Kreal][128] (zero-pad K)
__device__ __forceinline__ void fill_wswz(_Float16* wswz, const float* __restrict__ Wg,
                                          int Kreal, int Ksteps, int tid, int nthr){
  const int total = Ksteps * 8 * 32 * 16;
  for (int idx = tid; idx < total; idx += nthr){
    const int i    = idx & 15;
    const int lane = (idx >> 4) & 31;
    const int rest = idx >> 9;
    const int nt   = rest & 7;
    const int kst  = rest >> 3;
    const int k = kst*32 + ((lane & 16) ? 8 : 0) + ((i & 8) << 1) + (i & 7);
    const int n = nt*16 + (lane & 15);
    const float w = (k < Kreal) ? Wg[(size_t)k*HD + n] : 0.0f;
    wswz[idx] = (_Float16)w;
  }
}

// ---------------- TDM: async 2D tile (rows x 256B) global -> padded LDS ----------------
#if __has_builtin(__builtin_amdgcn_tensor_load_to_lds) && __has_builtin(__builtin_amdgcn_s_wait_tensorcnt)
#define HAVE_TDM 1
typedef __attribute__((ext_vector_type(4))) unsigned tdm_u4;
typedef __attribute__((ext_vector_type(8))) int      tdm_i8;
typedef __attribute__((ext_vector_type(4))) int      tdm_i4;

// Loads `rows` rows of 64 DWORDs (256 B) each from a [NEDGES][64-DW] row-major
// tensor into LDS, inserting 4 DWORDs (16 B) of pad after every 64 DWORDs
// -> LDS row stride 272 B == 136 f16 (matches a_frag_h staging layout).
__device__ __forceinline__ void tdm_load_rows(const void* gsrc, const void* lds_dst, int rows){
  const unsigned long long ga = (unsigned long long)gsrc;
  const unsigned lofs = (unsigned)(unsigned long long)lds_dst; // low 32 bits = LDS offset
  tdm_u4 g0;
  g0[0] = 1u;                                              // count=1 (valid user D#)
  g0[1] = lofs;                                            // lds_addr
  g0[2] = (unsigned)(ga & 0xFFFFFFFFull);                  // global_addr[31:0]
  g0[3] = (unsigned)((ga >> 32) & 0x1FFFFFFull) | (2u<<30);// global_addr[56:32], type=2
  tdm_i8 g1;
  g1[0] = (2 << 16)      // data_size = 4 B
        | (1 << 20)      // pad_enable
        | (5 << 22)      // pad_interval: code 5 = every 64 DWORDs
        | (3 << 25);     // pad_amount:   code 3 = 4 DWORDs
  g1[1] = (64 << 16);                        // tensor_dim0 = 64 (bits [79:48], low half)
  g1[2] = (int)((NEDGES & 0xFFFF) << 16);    // tensor_dim1[15:0] at [95:80]
  g1[3] = (NEDGES >> 16) | (64 << 16);       // tensor_dim1[31:16] | tile_dim0 = 64
  g1[4] = rows;                              // tile_dim1 (tile_dim2 = 0)
  g1[5] = 64;                                // tensor_dim0_stride = 64 DWORDs
  g1[6] = 0;
  g1[7] = 0;
  const tdm_i4 z4 = {0,0,0,0};
  const tdm_i8 z8 = {0,0,0,0,0,0,0,0};
  // 6-arg form (amdgpu-toolchain / clang-23): (g0, g1, g2, g3, g4, cpol)
  __builtin_amdgcn_tensor_load_to_lds(g0, g1, z4, z4, z8, 0);
}
#else
#define HAVE_TDM 0
#endif

// ---------------- init: V = emb_nodes @ pre_Wn + pre_bn ----------------
__global__ __launch_bounds__(256) void init_v_kernel(const float* __restrict__ emb_nodes,
    const float* __restrict__ Wn, const float* __restrict__ bn, float* __restrict__ V){
  for (int idx = blockIdx.x*blockDim.x + threadIdx.x; idx < NNODES*HD; idx += gridDim.x*blockDim.x){
    const int i = idx >> 7, c = idx & 127;
    V[idx] = emb_nodes[i*3+0]*Wn[0*HD+c] + emb_nodes[i*3+1]*Wn[1*HD+c]
           + emb_nodes[i*3+2]*Wn[2*HD+c] + bn[c];
  }
}

// ---------------- EK1: t1 = silu([V_src,V_dst,dist,emb_e] @ We1 + be1) ----------------
__global__ __launch_bounds__(512) void ek1_kernel(
    const float* __restrict__ V, const float* __restrict__ coords,
    const float* __restrict__ emb_edges, const int* __restrict__ ei,
    const float* __restrict__ We1, const float* __restrict__ be1,
    _Float16* __restrict__ t1)
{
  constexpr int KS = 9;        // ceil(259/32) -> K padded to 288
  constexpr int HSTR = 296;    // padded LDS row stride (f16)
  __shared__ alignas(32) _Float16 wswz[KS*8*32*16];   // 73728 B
  __shared__ float bias[HD];
  __shared__ alignas(16) _Float16 hbuf[32*HSTR];      // 18944 B

  const int tid = threadIdx.x;
  fill_wswz(wswz, We1, 259, KS, tid, 512);
  if (tid < HD) bias[tid] = be1[tid];
  __syncthreads();

  const int lane = tid & 31, wave = tid >> 5;
  const int mt = wave >> 3, nt = wave & 7;           // 2 row-tiles x 8 col-tiles
  const int col  = nt*16 + (lane & 15);
  const int rsub = (lane & 16) ? 8 : 0;
  const float bb = bias[col];

  for (int tile = blockIdx.x; tile < NEDGES/32; tile += gridDim.x){
    const int e0 = tile*32;
    { // stage h-tile: 16 threads per edge
      const int le = tid >> 4, p = tid & 15;
      const int e = e0 + le;
      const int s = ei[e], d = ei[NEDGES + e];
      _Float16* hr = hbuf + le*HSTR;
      const float* vs = V + (size_t)s*HD;
      const float* vd = V + (size_t)d*HD;
#pragma unroll
      for (int j = 0; j < 8; ++j){
        const int c = p*8 + j;
        hr[c]      = (_Float16)vs[c];
        hr[HD + c] = (_Float16)vd[c];
      }
      if (p == 0){
        const float dx = coords[d*3+0]-coords[s*3+0];
        const float dy = coords[d*3+1]-coords[s*3+1];
        const float dz = coords[d*3+2]-coords[s*3+2];
        hr[256] = (_Float16)sqrtf(dx*dx+dy*dy+dz*dz);
        hr[257] = (_Float16)emb_edges[(size_t)e*2+0];
        hr[258] = (_Float16)emb_edges[(size_t)e*2+1];
        for (int c = 259; c < 288; ++c) hr[c] = (_Float16)0.0f;
      }
    }
    __syncthreads();
    v8f acc = {0.0f,0.0f,0.0f,0.0f,0.0f,0.0f,0.0f,0.0f};
#pragma unroll
    for (int ks = 0; ks < KS; ++ks){
      v16h a = a_frag_h(hbuf + mt*16*HSTR, HSTR, ks*32, lane);
      v16h b = b_frag(wswz, ks, nt, lane);
      acc = __builtin_amdgcn_wmma_f32_16x16x32_f16(false, a, false, b, (short)0, acc, false, false);
    }
    const int rbase = e0 + mt*16 + rsub;
#pragma unroll
    for (int r = 0; r < 8; ++r)
      t1[(size_t)(rbase + r)*HD + col] = (_Float16)silu_f(acc[r] + bb);
    __syncthreads();
  }
}

// ---- EK2: mij = silu(t1@We2+be2); t2 = silu(mij@Wx1+bx1); px = t2@Wx2+bx2;
//           atomics: coord_acc[dst]+=diff*px, mi[dst]+=mij; last: Eact=elu(mij)+BN sums
__global__ __launch_bounds__(512) void ek2_kernel(
    const _Float16* __restrict__ t1, const float* __restrict__ coords,
    const int* __restrict__ ei,
    const float* __restrict__ We2, const float* __restrict__ be2,
    const float* __restrict__ Wx1, const float* __restrict__ bx1,
    const float* __restrict__ Wx2, const float* __restrict__ bx2,
    float* __restrict__ coord_acc, float* __restrict__ mi,
    float* __restrict__ Eout, float* __restrict__ bn_e, int last)
{
  __shared__ alignas(32) _Float16 wswz2[4*8*32*16];   // 32768 B
  __shared__ alignas(32) _Float16 wswzx[4*8*32*16];   // 32768 B
  __shared__ float bias2[HD], biasx[HD];
  __shared__ float wx2s[HD*3];
  __shared__ float bx2s[4];
  __shared__ alignas(16) _Float16 t1buf[2][32*136];   // 2 x 8704 B (double buffer)
  __shared__ alignas(16) float mijb[32*132];          // 16896 B
  __shared__ alignas(16) float t2b[32*132];           // 16896 B
  __shared__ float part[3*32*16];                     // 6144 B

  const int tid = threadIdx.x;
  fill_wswz(wswz2, We2, 128, 4, tid, 512);
  fill_wswz(wswzx, Wx1, 128, 4, tid, 512);
  if (tid < HD){ bias2[tid] = be2[tid]; biasx[tid] = bx1[tid]; }
  for (int i = tid; i < HD*3; i += 512) wx2s[i] = Wx2[i];
  if (tid < 3) bx2s[tid] = bx2[tid];

  const int lane = tid & 31, wave = tid >> 5;
  const int mt = wave >> 3, nt = wave & 7;
  const int col = nt*16 + (lane & 15);
  const int rowoff = mt*16 + ((lane & 16) ? 8 : 0);
  const int NT = NEDGES/32;

#if HAVE_TDM
  if (wave == 0 && blockIdx.x < NT)
    tdm_load_rows(t1 + (size_t)blockIdx.x*32*HD, &t1buf[0][0], 32);
#endif
  __syncthreads();
  const float b2 = bias2[col], bx = biasx[col];

  int it = 0;
  for (int tile = blockIdx.x; tile < NT; tile += gridDim.x, ++it){
    const int e0 = tile*32;
    const int cur = it & 1;
#if HAVE_TDM
    if (wave == 0) __builtin_amdgcn_s_wait_tensorcnt(0);
    __syncthreads();                 // tile data in t1buf[cur] visible to all waves
    if (wave == 0){                  // prefetch next tile while we compute
      const int nxt = tile + gridDim.x;
      if (nxt < NT) tdm_load_rows(t1 + (size_t)nxt*32*HD, &t1buf[cur^1][0], 32);
    }
#else
    { // stage t1 tile (coalesced fallback)
      const int le = tid >> 4, c0 = (tid & 15) * 8;
      const _Float16* src = t1 + (size_t)(e0 + le)*HD + c0;
      _Float16* dstp = &t1buf[cur][0] + le*136 + c0;
#pragma unroll
      for (int j = 0; j < 8; ++j) dstp[j] = src[j];
    }
    __syncthreads();
#endif
    { // GEMM2 -> mij (f32 in LDS)
      v8f acc = {0.0f,0.0f,0.0f,0.0f,0.0f,0.0f,0.0f,0.0f};
#pragma unroll
      for (int ks = 0; ks < 4; ++ks){
        v16h a = a_frag_h(&t1buf[cur][0] + mt*16*136, 136, ks*32, lane);
        v16h b = b_frag(wswz2, ks, nt, lane);
        acc = __builtin_amdgcn_wmma_f32_16x16x32_f16(false, a, false, b, (short)0, acc, false, false);
      }
#pragma unroll
      for (int r = 0; r < 8; ++r)
        mijb[(rowoff + r)*132 + col] = silu_f(acc[r] + b2);
    }
    __syncthreads();
    { // GEMM3 -> t2 (f32 in LDS)
      v8f acc = {0.0f,0.0f,0.0f,0.0f,0.0f,0.0f,0.0f,0.0f};
#pragma unroll
      for (int ks = 0; ks < 4; ++ks){
        v16h a = a_frag_f(mijb + mt*16*132, 132, ks*32, lane);
        v16h b = b_frag(wswzx, ks, nt, lane);
        acc = __builtin_amdgcn_wmma_f32_16x16x32_f16(false, a, false, b, (short)0, acc, false, false);
      }
#pragma unroll
      for (int r = 0; r < 8; ++r)
        t2b[(rowoff + r)*132 + col] = silu_f(acc[r] + bx);
    }
    __syncthreads();
    { // px partial dot products
      const int le = tid >> 4, p = tid & 15;
      float s0 = 0.f, s1 = 0.f, s2 = 0.f;
      const float* t2r = t2b + le*132;
#pragma unroll
      for (int j = 0; j < 8; ++j){
        const int c = p*8 + j;
        const float v = t2r[c];
        s0 += v * wx2s[c*3+0];
        s1 += v * wx2s[c*3+1];
        s2 += v * wx2s[c*3+2];
      }
      part[0*512 + le*16 + p] = s0;
      part[1*512 + le*16 + p] = s1;
      part[2*512 + le*16 + p] = s2;
    }
    __syncthreads();
    if (tid < 32){ // finalize px; coord_acc atomics
      const int le = tid;
      float p0 = bx2s[0], p1 = bx2s[1], p2 = bx2s[2];
      for (int p = 0; p < 16; ++p){
        p0 += part[0*512 + le*16 + p];
        p1 += part[1*512 + le*16 + p];
        p2 += part[2*512 + le*16 + p];
      }
      const int e = e0 + le;
      const int s = ei[e], d = ei[NEDGES + e];
      const float dx = coords[d*3+0]-coords[s*3+0];
      const float dy = coords[d*3+1]-coords[s*3+1];
      const float dz = coords[d*3+2]-coords[s*3+2];
      atomicAdd(&coord_acc[d*3+0], dx*p0);
      atomicAdd(&coord_acc[d*3+1], dy*p1);
      atomicAdd(&coord_acc[d*3+2], dz*p2);
    }
    { // mi[dst] += mij
      const int le = tid >> 4, p = tid & 15;
      const int d = ei[NEDGES + e0 + le];
      float* dstm = mi + (size_t)d*HD;
      const float* srcm = mijb + le*132;
#pragma unroll
      for (int j = 0; j < 8; ++j){
        const int c = p*8 + j;
        atomicAdd(dstm + c, srcm[c]);
      }
    }
    if (last && tid < HD){ // Eact = elu(mij) -> Eout, BN partial sums
      const int c = tid;
      float s = 0.f, sq = 0.f;
      for (int le = 0; le < 32; ++le){
        const float x = elu_f(mijb[le*132 + c]);
        Eout[(size_t)(e0 + le)*HD + c] = x;
        s += x; sq += x*x;
      }
      atomicAdd(&bn_e[c], s);
      atomicAdd(&bn_e[HD + c], sq);
    }
    __syncthreads();
  }
}

// ---- NK1a: coords += coord_acc/(N-1);  t = silu([V,mi]@Wh1+bh1) (f16 out) ----
__global__ __launch_bounds__(256) void nk1a_kernel(
    const float* __restrict__ V, const float* __restrict__ mi,
    const float* __restrict__ Wh1, const float* __restrict__ bh1,
    const float* __restrict__ coord_acc, float* __restrict__ coords,
    _Float16* __restrict__ tbuf)
{
  __shared__ alignas(32) _Float16 wswz[8*8*32*16];    // 65536 B
  __shared__ float bias[HD];
  __shared__ alignas(16) _Float16 abuf[16*264];       // 8448 B
  const int tid = threadIdx.x;
  fill_wswz(wswz, Wh1, 256, 8, tid, 256);
  if (tid < HD) bias[tid] = bh1[tid];
  __syncthreads();

  const int lane = tid & 31, wave = tid >> 5;         // 8 waves = 8 col-tiles
  const int col = wave*16 + (lane & 15);
  const int rsub = (lane & 16) ? 8 : 0;
  const float bb = bias[col];

  for (int tile = blockIdx.x; tile < NNODES/16; tile += gridDim.x){
    const int i0 = tile*16;
    if (tid < 48){ // equivariant coordinate update (done exactly once per node)
      const int i = i0 + tid/3, cp = tid % 3;
      coords[i*3+cp] += coord_acc[i*3+cp] * (1.0f/(float)(NNODES-1));
    }
    { // stage [V | mi] tile as f16
      const int le = tid >> 4, c0 = (tid & 15)*16;
      _Float16* dr = abuf + le*264 + c0;
      const int i = i0 + le;
#pragma unroll
      for (int j = 0; j < 16; ++j){
        const int c = c0 + j;
        const float v = (c < HD) ? V[(size_t)i*HD + c] : mi[(size_t)i*HD + (c - HD)];
        dr[j] = (_Float16)v;
      }
    }
    __syncthreads();
    v8f acc = {0.0f,0.0f,0.0f,0.0f,0.0f,0.0f,0.0f,0.0f};
#pragma unroll
    for (int ks = 0; ks < 8; ++ks){
      v16h a = a_frag_h(abuf, 264, ks*32, lane);
      v16h b = b_frag(wswz, ks, wave, lane);
      acc = __builtin_amdgcn_wmma_f32_16x16x32_f16(false, a, false, b, (short)0, acc, false, false);
    }
#pragma unroll
    for (int r = 0; r < 8; ++r)
      tbuf[(size_t)(i0 + rsub + r)*HD + col] = (_Float16)silu_f(acc[r] + bb);
    __syncthreads();
  }
}

// ---- NK1b: Vact = elu(t@Wh2+bh2); BN partial sums ----
__global__ __launch_bounds__(256) void nk1b_kernel(
    const _Float16* __restrict__ tbuf,
    const float* __restrict__ Wh2, const float* __restrict__ bh2,
    float* __restrict__ Vact, float* __restrict__ bn_n)
{
  __shared__ alignas(32) _Float16 wswz[4*8*32*16];    // 32768 B
  __shared__ float bias[HD];
  __shared__ alignas(16) _Float16 abuf[16*136];       // 4352 B
  const int tid = threadIdx.x;
  fill_wswz(wswz, Wh2, 128, 4, tid, 256);
  if (tid < HD) bias[tid] = bh2[tid];
  __syncthreads();

  const int lane = tid & 31, wave = tid >> 5;
  const int col = wave*16 + (lane & 15);
  const int rsub = (lane & 16) ? 8 : 0;
  const float bb = bias[col];

  for (int tile = blockIdx.x; tile < NNODES/16; tile += gridDim.x){
    const int i0 = tile*16;
    {
      const int le = tid >> 4, c0 = (tid & 15)*8;
      const _Float16* src = tbuf + (size_t)(i0 + le)*HD + c0;
      _Float16* dr = abuf + le*136 + c0;
#pragma unroll
      for (int j = 0; j < 8; ++j) dr[j] = src[j];
    }
    __syncthreads();
    v8f acc = {0.0f,0.0f,0.0f,0.0f,0.0f,0.0f,0.0f,0.0f};
#pragma unroll
    for (int ks = 0; ks < 4; ++ks){
      v16h a = a_frag_h(abuf, 136, ks*32, lane);
      v16h b = b_frag(wswz, ks, wave, lane);
      acc = __builtin_amdgcn_wmma_f32_16x16x32_f16(false, a, false, b, (short)0, acc, false, false);
    }
    float s = 0.f, sq = 0.f;
#pragma unroll
    for (int r = 0; r < 8; ++r){
      const float x = elu_f(acc[r] + bb);
      Vact[(size_t)(i0 + rsub + r)*HD + col] = x;
      s += x; sq += x*x;
    }
    atomicAdd(&bn_n[col], s);
    atomicAdd(&bn_n[HD + col], sq);
    __syncthreads();
  }
}

// ---- BN finalize + apply ----
__global__ void bnfin_kernel(const float* __restrict__ bn_n, float* __restrict__ stat_n,
                             const float* __restrict__ bn_e, float* __restrict__ stat_e, int last){
  const int t = threadIdx.x;
  if (t < HD){
    const float mu = bn_n[t] * (1.0f/(float)NNODES);
    const float var = bn_n[HD+t] * (1.0f/(float)NNODES) - mu*mu;
    stat_n[t] = mu;
    stat_n[HD+t] = rsqrtf(var + 1e-5f);
    if (last){
      const float mue = bn_e[t] * (1.0f/(float)NEDGES);
      const float vare = bn_e[HD+t] * (1.0f/(float)NEDGES) - mue*mue;
      stat_e[t] = mue;
      stat_e[HD+t] = rsqrtf(vare + 1e-5f);
    }
  }
}

__global__ __launch_bounds__(256) void bn_apply_kernel(
    const float* __restrict__ src, float* __restrict__ dst,
    const float* __restrict__ stat, const float* __restrict__ gam,
    const float* __restrict__ bet, int rows){
  const long total = (long)rows * HD;
  for (long idx = (long)blockIdx.x*blockDim.x + threadIdx.x; idx < total;
       idx += (long)gridDim.x*blockDim.x){
    const int c = (int)(idx & (HD-1));
    dst[idx] = gam[c] * (src[idx] - stat[c]) * stat[HD + c] + bet[c];
  }
}

// ---------------- host launcher ----------------
extern "C" void kernel_launch(void* const* d_in, const int* in_sizes, int n_in,
                              void* d_out, int out_size, void* d_ws, size_t ws_size,
                              hipStream_t stream) {
  (void)in_sizes; (void)n_in; (void)out_size; (void)ws_size;
  const float* emb_nodes = (const float*)d_in[0];
  const float* emb_edges = (const float*)d_in[1];
  const int*   ei        = (const int*)d_in[2];
  const float* pre_Wn = (const float*)d_in[3];
  const float* pre_bn = (const float*)d_in[4];
  // d_in[5]/d_in[6] (pre_We/pre_be) are dead in the reference (E is overwritten)
  const float* We1 = (const float*)d_in[7];
  const float* be1 = (const float*)d_in[8];
  const float* We2 = (const float*)d_in[9];
  const float* be2 = (const float*)d_in[10];
  const float* Wx1 = (const float*)d_in[11];
  const float* bx1 = (const float*)d_in[12];
  const float* Wx2 = (const float*)d_in[13];
  const float* bx2 = (const float*)d_in[14];
  const float* Wh1 = (const float*)d_in[15];
  const float* bh1 = (const float*)d_in[16];
  const float* Wh2 = (const float*)d_in[17];
  const float* bh2 = (const float*)d_in[18];
  const float* gam_n = (const float*)d_in[19];
  const float* bet_n = (const float*)d_in[20];
  const float* gam_e = (const float*)d_in[21];
  const float* bet_e = (const float*)d_in[22];

  float* outV = (float*)d_out;
  float* outE = (float*)d_out + (size_t)NNODES*HD;

  // workspace carve (256-B aligned bump allocator)
  char* wp = (char*)d_ws;
  auto carve = [&](size_t bytes)->void*{
    void* p = (void*)wp;
    wp += (bytes + 255) & ~(size_t)255;
    return p;
  };
  float*    coords    = (float*)carve((size_t)NNODES*3*sizeof(float));
  float*    coord_acc = (float*)carve((size_t)NNODES*3*sizeof(float));
  float*    V         = (float*)carve((size_t)NNODES*HD*sizeof(float));
  float*    mi        = (float*)carve((size_t)NNODES*HD*sizeof(float));
  float*    Vact      = (float*)carve((size_t)NNODES*HD*sizeof(float));
  _Float16* t1        = (_Float16*)carve((size_t)NEDGES*HD*sizeof(_Float16));
  float*    bn_n      = (float*)carve(256*sizeof(float));
  float*    bn_e      = (float*)carve(256*sizeof(float));
  float*    stat_n    = (float*)carve(256*sizeof(float));
  float*    stat_e    = (float*)carve(256*sizeof(float));

  (void)hipMemcpyAsync(coords, emb_nodes, (size_t)NNODES*3*sizeof(float),
                       hipMemcpyDeviceToDevice, stream);
  init_v_kernel<<<1250, 256, 0, stream>>>(emb_nodes, pre_Wn, pre_bn, V);

  for (int l = 0; l < 3; ++l){
    const int last = (l == 2);
    (void)hipMemsetAsync(coord_acc, 0, (size_t)NNODES*3*sizeof(float), stream);
    (void)hipMemsetAsync(mi, 0, (size_t)NNODES*HD*sizeof(float), stream);
    (void)hipMemsetAsync(bn_n, 0, 256*sizeof(float), stream);
    if (last) (void)hipMemsetAsync(bn_e, 0, 256*sizeof(float), stream);

    ek1_kernel<<<1250, 512, 0, stream>>>(V, coords, emb_edges, ei,
        We1 + (size_t)l*259*HD, be1 + l*HD, t1);
    ek2_kernel<<<1250, 512, 0, stream>>>(t1, coords, ei,
        We2 + (size_t)l*HD*HD, be2 + l*HD,
        Wx1 + (size_t)l*HD*HD, bx1 + l*HD,
        Wx2 + (size_t)l*HD*3,  bx2 + l*3,
        coord_acc, mi, outE, bn_e, last);
    nk1a_kernel<<<625, 256, 0, stream>>>(V, mi,
        Wh1 + (size_t)l*2*HD*HD, bh1 + l*HD, coord_acc, coords, t1);
    nk1b_kernel<<<625, 256, 0, stream>>>(t1,
        Wh2 + (size_t)l*HD*HD, bh2 + l*HD, Vact, bn_n);
    bnfin_kernel<<<1, 128, 0, stream>>>(bn_n, stat_n, bn_e, stat_e, last);
    bn_apply_kernel<<<1024, 256, 0, stream>>>(Vact, last ? outV : V,
        stat_n, gam_n + l*HD, bet_n + l*HD, NNODES);
    if (last)
      bn_apply_kernel<<<4096, 256, 0, stream>>>(outE, outE,
          stat_e, gam_e + 2*HD, bet_e + 2*HD, NEDGES);
  }
}
